// QuantumWalkRetriever_86543591014914
// MI455X (gfx1250) — matrix operand order repaired and checked by reading, not compiled, not run on hardware
//
#include <hip/hip_runtime.h>
#include <hip/hip_fp16.h>

// ---------------------------------------------------------------------------
// QuantumWalkRetriever for MI455X (gfx1250, wave32, WMMA).
//
// Phase 1 (dominant, 4.9 GFLOP):  h = relu(emb @ W1a + c),  amps = h @ W2 + b2
//   done as f16 WMMA (v_wmma_f32_16x16x32_f16) with f32 accumulate.
//   qv's contribution is a constant vector c folded in a prep kernel.
//   Each wave owns a 32-row macro-tile (two 16-row A fragments) so every
//   B fragment load feeds two WMMAs -> 1 vmem per wmma in steady state.
// Phase 2: reverse-slot lookup over nbr (gather; nbr fits in 192MB L2).
// Phase 3: 3 rank-1 walk steps (coin@s == b*(b.s) after folding both
//   normalizations into b), scatter-add + global L2 norm per step.
//
// WMMA operand layouts (CDNA5 ISA 7.12.2):
//   A (16x32 f16): lane<16 holds row M=lane, elems 0..7 = K0..7, 8..15 = K16..23;
//                  lane>=16 holds row M=lane-16, K8..15 then K24..31.
//   B (32x16 f16): same K order as A; lane holds column
//                  N=lane%16; lanes 0-15 K=0..7,16..23, lanes 16-31 K=8..15,24..31.
//                  W1/W2 are pre-swizzled so each B fragment is a contiguous
//                  32-byte per-lane load (2x global_load_b128, L2-resident).
//   C/D (16x16 f32): vgpr m + 8*(lane/16) = row, lane%16 = col.
// ---------------------------------------------------------------------------

typedef __attribute__((ext_vector_type(16))) _Float16 v16h;
typedef __attribute__((ext_vector_type(8)))  float    v8f;

#define NN 50000
#define DD 384
#define HH 128
#define KK 16
#define EPSF 1e-8f
#define KSLICES 12           // DD / 32
#define CTILES  8            // HH / 16
#define GWAVES  4            // waves (32-row macro-tiles) per block
#define MTILES  1563         // ceil(NN / 32); last macro-tile upper half invalid
#define LDS_STRIDE 132       // 128 + 4 f16 pad -> 66-dword row stride, conflict-free

// ---------------- prep kernels -------------------------------------------

__global__ void prep_cvec(const float* __restrict__ qv, const float* __restrict__ W1,
                          const float* __restrict__ b1, float* __restrict__ cvec) {
    int h = threadIdx.x;   // 128 threads
    float acc = b1[h];
    for (int d = 0; d < DD; ++d) acc += qv[d] * W1[(size_t)(DD + d) * HH + h];
    cvec[h] = acc;
}

// W1s[((c*12+s)*32+lane)*16+j] = f16( W1[(s*32 + (lane/16)*8 + (j%8) + (j/8)*16)*128 + c*16 + lane%16] )
__global__ void prep_w1s(const float* __restrict__ W1, _Float16* __restrict__ W1s) {
    int t = blockIdx.x * blockDim.x + threadIdx.x;
    if (t >= CTILES * KSLICES * 32 * 16) return;
    int j    = t & 15;
    int lane = (t >> 4) & 31;
    int s    = (t >> 9) % KSLICES;
    int c    = t / (KSLICES * 512);
    int k    = s * 32 + (lane >> 4) * 8 + (j & 7) + (j >> 3) * 16;
    int col  = c * 16 + (lane & 15);
    W1s[t] = (_Float16)W1[(size_t)k * HH + col];
}

__global__ void prep_w2s(const float* __restrict__ W2, _Float16* __restrict__ W2s) {
    int t = blockIdx.x * blockDim.x + threadIdx.x;
    if (t >= 4 * 32 * 16) return;
    int j    = t & 15;
    int lane = (t >> 4) & 31;
    int s2   = t >> 9;
    int k    = s2 * 32 + (lane >> 4) * 8 + (j & 7) + (j >> 3) * 16;
    W2s[t] = (_Float16)W2[(size_t)k * KK + (lane & 15)];
}

// ---------------- fused MLP + coin-vector kernel (WMMA) -------------------

__global__ __launch_bounds__(GWAVES * 32)
void gemm_coin_kernel(const float* __restrict__ emb,
                      const _Float16* __restrict__ W1s,
                      const _Float16* __restrict__ W2s,
                      const float* __restrict__ cvec,
                      const float* __restrict__ b2,
                      float* __restrict__ bvec) {
    __shared__ _Float16 lds_h[GWAVES * 32 * LDS_STRIDE];

    const int w    = threadIdx.x >> 5;
    const int lane = threadIdx.x & 31;
    const int hf   = lane >> 4;     // lane half
    const int n16  = lane & 15;
    const int tile = blockIdx.x * GWAVES + w;
    if (tile >= MTILES) return;     // no barriers used -> safe early exit
    const int row0 = tile * 32;

    v8f acc0[CTILES], acc1[CTILES];
#pragma unroll
    for (int c = 0; c < CTILES; ++c) { acc0[c] = {}; acc1[c] = {}; }

    // Lane owns row row0+n16 (frag 0) and row0+16+n16 (frag 1, clamped: WMMA
    // needs EXEC all-ones, so out-of-range rows compute on clamped data and
    // only the stores are guarded).
    int rB = row0 + 16 + n16;
    if (rB >= NN) rB = 0;
    const float2* erowA = (const float2*)(emb + (size_t)(row0 + n16) * DD);
    const float2* erowB = (const float2*)(emb + (size_t)rB * DD);

#pragma unroll 1
    for (int s = 0; s < KSLICES; ++s) {
        const int kb = s * 32 + hf * 8;     // element (f32) index base, even
        v16h a0, a1;
#pragma unroll
        for (int i = 0; i < 4; ++i) {
            float2 f0 = erowA[kb / 2 + i];       // K = kb + 2i, kb + 2i + 1
            float2 f1 = erowA[kb / 2 + 8 + i];   // K = kb + 16 + 2i, ...
            a0[2 * i]     = (_Float16)f0.x;
            a0[2 * i + 1] = (_Float16)f0.y;
            a0[8 + 2 * i] = (_Float16)f1.x;
            a0[9 + 2 * i] = (_Float16)f1.y;
            float2 g0 = erowB[kb / 2 + i];
            float2 g1 = erowB[kb / 2 + 8 + i];
            a1[2 * i]     = (_Float16)g0.x;
            a1[2 * i + 1] = (_Float16)g0.y;
            a1[8 + 2 * i] = (_Float16)g1.x;
            a1[9 + 2 * i] = (_Float16)g1.y;
        }
#pragma unroll
        for (int c = 0; c < CTILES; ++c) {
            const v16h b = *(const v16h*)(W1s + ((size_t)(c * KSLICES + s) * 32 + lane) * 16);
            acc0[c] = __builtin_amdgcn_wmma_f32_16x16x32_f16(
                false, a0, false, b, (short)0, acc0[c], false, false);
            acc1[c] = __builtin_amdgcn_wmma_f32_16x16x32_f16(
                false, a1, false, b, (short)0, acc1[c], false, false);
        }
    }

    // bias + relu -> per-wave-private LDS (f16, row-major, padded, 32 rows)
    _Float16* lw = lds_h + w * 32 * LDS_STRIDE;
#pragma unroll
    for (int c = 0; c < CTILES; ++c) {
        const int col = c * 16 + n16;
        const float cb = cvec[col];
#pragma unroll
        for (int m = 0; m < 8; ++m) {
            float hv0 = acc0[c][m] + cb;
            hv0 = hv0 > 0.0f ? hv0 : 0.0f;
            lw[(m + 8 * hf) * LDS_STRIDE + col] = (_Float16)hv0;
            float hv1 = acc1[c][m] + cb;
            hv1 = hv1 > 0.0f ? hv1 : 0.0f;
            lw[(16 + m + 8 * hf) * LDS_STRIDE + col] = (_Float16)hv1;
        }
    }

    // amps tile = h(16x128) @ W2(128x16): 4 accumulating WMMAs per 16-row tile
    const float b2v = b2[n16];
#pragma unroll
    for (int t2 = 0; t2 < 2; ++t2) {
        v8f accA = {};
#pragma unroll
        for (int s2 = 0; s2 < 4; ++s2) {
            const _Float16* lp = lw + (t2 * 16 + n16) * LDS_STRIDE + s2 * 32 + hf * 8;
            union { v16h v; unsigned u[8]; } au;
#pragma unroll
            for (int i = 0; i < 4; ++i) {
                au.u[i]     = *(const unsigned*)(lp + 2 * i);       // K pairs 0..7 (local)
                au.u[4 + i] = *(const unsigned*)(lp + 16 + 2 * i);  // K pairs 16..23 (local)
            }
            const v16h b = *(const v16h*)(W2s + ((size_t)(s2 * 32) + lane) * 16);
            accA = __builtin_amdgcn_wmma_f32_16x16x32_f16(
                false, au.v, false, b, (short)0, accA, false, false);
        }
        // amps += b2; normalize; fold Frobenius normalization of the rank-1
        // coin into one vector: coin@s == bv * (bv . s), bv = a*rsqrt(fro+eps).
#pragma unroll
        for (int m = 0; m < 8; ++m) {
            float x = accA[m] + b2v;           // row = t2*16 + m + 8*hf, col = n16
            float sq = x * x;                  // reduce across the 16-lane half
            sq += __shfl_xor(sq, 1, 32);
            sq += __shfl_xor(sq, 2, 32);
            sq += __shfl_xor(sq, 4, 32);
            sq += __shfl_xor(sq, 8, 32);
            float nrm = sqrtf(sq) + EPSF;
            float an  = x / nrm;               // normalized amplitude
            float fro = sq / (nrm * nrm);      // = sum(an^2) = Frobenius of outer
            float bv  = an * rsqrtf(fro + EPSF);
            int row = row0 + t2 * 16 + m + 8 * hf;
            if (row < NN) bvec[(size_t)row * KK + n16] = bv;
        }
    }
}

// ---------------- reverse-slot lookup -------------------------------------

__global__ void rev_kernel(const int* __restrict__ nbr, int* __restrict__ tgt) {
    int e = blockIdx.x * blockDim.x + threadIdx.x;
    if (e >= NN * KK) return;
    int i = e >> 4;
    int j = nbr[e];
    const int* rowj = nbr + (size_t)j * KK;
    int t = -1;
#pragma unroll
    for (int p = KK - 1; p >= 0; --p)     // descending overwrite -> first match
        if (rowj[p] == i) t = j * KK + p;
    tgt[e] = t;
}

// ---------------- walk kernels --------------------------------------------

__global__ void init_kernel(float* __restrict__ st) {
    int e = blockIdx.x * blockDim.x + threadIdx.x;
    if (e < NN * KK) st[e] = rsqrtf((float)(NN * KK));
}

__global__ void zero_kernel(float* __restrict__ nxt, float* __restrict__ nrm) {
    int e = blockIdx.x * blockDim.x + threadIdx.x;
    if (e < NN * KK) nxt[e] = 0.0f;
    if (e == 0) *nrm = 0.0f;
}

__global__ void scatter_kernel(const float* __restrict__ bvec,
                               const float* __restrict__ cur,
                               const int* __restrict__ tgt,
                               float* __restrict__ nxt,
                               const float* __restrict__ prevnorm) {
    int i = blockIdx.x * blockDim.x + threadIdx.x;
    if (i >= NN) return;
    float s = 1.0f;
    if (prevnorm) s = 1.0f / (sqrtf(*prevnorm) + EPSF);  // deferred normalization
    const float4* bp = (const float4*)(bvec + (size_t)i * KK);
    const float4* cp = (const float4*)(cur + (size_t)i * KK);
    float bl[KK];
    float dot = 0.0f;
#pragma unroll
    for (int q = 0; q < 4; ++q) {
        float4 bb = bp[q], cc = cp[q];
        bl[4 * q + 0] = bb.x; bl[4 * q + 1] = bb.y;
        bl[4 * q + 2] = bb.z; bl[4 * q + 3] = bb.w;
        dot += bb.x * cc.x + bb.y * cc.y + bb.z * cc.z + bb.w * cc.w;
    }
    dot *= s;
    const int* trow = tgt + (size_t)i * KK;
#pragma unroll
    for (int k = 0; k < KK; ++k) {
        int t = trow[k];
        if (t >= 0) unsafeAtomicAdd(nxt + t, bl[k] * dot);
    }
}

__global__ void norm_kernel(const float* __restrict__ v, float* __restrict__ out) {
    int e = blockIdx.x * blockDim.x + threadIdx.x;
    float x = (e < NN * KK) ? v[e] : 0.0f;
    float sq = x * x;
#pragma unroll
    for (int d = 1; d < 32; d <<= 1) sq += __shfl_xor(sq, d, 32);
    if ((threadIdx.x & 31) == 0) unsafeAtomicAdd(out, sq);
}

__global__ void final_kernel(const float* __restrict__ st,
                             const float* __restrict__ nrm,
                             float* __restrict__ out) {
    int i = blockIdx.x * blockDim.x + threadIdx.x;
    if (i >= NN) return;
    float s = 1.0f / (sqrtf(*nrm) + EPSF);
    const float4* sp = (const float4*)(st + (size_t)i * KK);
    float p = 0.0f;
#pragma unroll
    for (int q = 0; q < 4; ++q) {
        float4 v = sp[q];
        float a = v.x * s, b = v.y * s, c = v.z * s, d = v.w * s;
        p += a * a + b * b + c * c + d * d;
    }
    if (isnan(p)) p = 0.0f;
    else if (isinf(p)) p = (p > 0.0f) ? 1.0f : 0.0f;
    out[i] = p;
}

// ---------------- launcher -------------------------------------------------

extern "C" void kernel_launch(void* const* d_in, const int* in_sizes, int n_in,
                              void* d_out, int out_size, void* d_ws, size_t ws_size,
                              hipStream_t stream) {
    const float* emb = (const float*)d_in[0];
    const float* qv  = (const float*)d_in[1];
    const float* W1  = (const float*)d_in[2];
    const float* b1  = (const float*)d_in[3];
    const float* W2  = (const float*)d_in[4];
    const float* b2  = (const float*)d_in[5];
    const int*   nbr = (const int*)d_in[6];
    float* out = (float*)d_out;

    // workspace layout (all offsets 16B-aligned; ~12.9 MB total)
    char* ws = (char*)d_ws;
    _Float16* W1s  = (_Float16*)(ws + 0);          //  98304 B
    _Float16* W2s  = (_Float16*)(ws + 98304);      //   4096 B
    float*    cvec = (float*)   (ws + 102400);     //    512 B
    float*    bvec = (float*)   (ws + 102912);     // 3200000 B
    int*      tgt  = (int*)     (ws + 3302912);    // 3200000 B
    float*    stA  = (float*)   (ws + 6502912);    // 3200000 B
    float*    stB  = (float*)   (ws + 9702912);    // 3200000 B
    float*    nrm  = (float*)   (ws + 12902912);   // 3 floats

    prep_cvec<<<1, 128, 0, stream>>>(qv, W1, b1, cvec);
    prep_w1s<<<(CTILES * KSLICES * 512 + 255) / 256, 256, 0, stream>>>(W1, W1s);
    prep_w2s<<<8, 256, 0, stream>>>(W2, W2s);

    gemm_coin_kernel<<<(MTILES + GWAVES - 1) / GWAVES, GWAVES * 32, 0, stream>>>(
        emb, W1s, W2s, cvec, b2, bvec);

    rev_kernel<<<(NN * KK + 255) / 256, 256, 0, stream>>>(nbr, tgt);
    init_kernel<<<(NN * KK + 255) / 256, 256, 0, stream>>>(stA);

    float* cur = stA;
    float* nx  = stB;
    for (int t = 0; t < 3; ++t) {
        zero_kernel<<<(NN * KK + 255) / 256, 256, 0, stream>>>(nx, nrm + t);
        scatter_kernel<<<(NN + 255) / 256, 256, 0, stream>>>(
            bvec, cur, tgt, nx, t ? (nrm + t - 1) : (const float*)nullptr);
        norm_kernel<<<(NN * KK + 255) / 256, 256, 0, stream>>>(nx, nrm + t);
        float* tmp = cur; cur = nx; nx = tmp;
    }
    final_kernel<<<(NN + 255) / 256, 256, 0, stream>>>(cur, nrm + 2, out);
}
